// RandLANet_34239479284330
// MI455X (gfx1250) — matrix operand order repaired and checked by reading, not compiled, not run on hardware
//
#include <hip/hip_runtime.h>
#include <hip/hip_bf16.h>
#include <math.h>

typedef __attribute__((ext_vector_type(16))) _Float16 v16h;
typedef __attribute__((ext_vector_type(8)))  _Float16 v8h;
typedef __attribute__((ext_vector_type(8)))  float    v8f;
typedef __attribute__((ext_vector_type(4)))  float    v4f;

#define DEVI __device__ __forceinline__

// ---------------------------------------------------------------------------
// WMMA fragment helpers (gfx1250 wave32 layouts per CDNA5 ISA 7.12.2)
// A (16x32 f16): lane l<16 -> row l, K = [k0..k0+8) and [k0+16..k0+24); l>=16 -> k0+=8
// B (32x16 f16): lane l    -> col l&15, K = [k0..k0+8) and [k0+16..k0+24); l>=16 -> k0+=8
// C/D (16x16 f32): lane l, vgpr v -> m = v + 8*(l>=16), n = l&15
// ---------------------------------------------------------------------------
DEVI v8f wz8() {
  v8f a;
#pragma unroll
  for (int i = 0; i < 8; ++i) a[i] = 0.f;
  return a;
}

DEVI v16h frag_a_lds(const _Float16* buf, int ld, int kbase, int lane) {
  const int m  = lane & 15;
  const int k0 = kbase + ((lane >> 4) << 3);
  v8h lo = *(const v8h*)(buf + m * ld + k0);
  v8h hi = *(const v8h*)(buf + m * ld + k0 + 16);
  v16h a;
#pragma unroll
  for (int i = 0; i < 8; ++i) { a[i] = lo[i]; a[8 + i] = hi[i]; }
  return a;
}

// Unchecked B fragment: valid only when all 16 tile rows exist and ks+24 <= K.
DEVI v16h frag_b_full(const _Float16* W, int Kphys, int kbase, int lane) {
  const int n  = lane & 15;
  const int ks = kbase + ((lane >> 4) << 3);
  v8h l0 = *(const v8h*)(W + (long)n * Kphys + ks);
  v8h l1 = *(const v8h*)(W + (long)n * Kphys + ks + 16);
  v16h b;
#pragma unroll
  for (int i = 0; i < 8; ++i) { b[i] = l0[i]; b[8 + i] = l1[i]; }
  return b;
}

// Checked B fragment (edge tiles / short K). W already offset to tile row 0.
DEVI v16h frag_b_gl(const _Float16* W, int Kphys, int Klimit, int kbase, int lane, int nmax) {
  const int n  = lane & 15;
  const int ks = kbase + ((lane >> 4) << 3);
  v16h b;
  if (n >= nmax || ks >= Klimit) {
#pragma unroll
    for (int i = 0; i < 16; ++i) b[i] = (_Float16)0.f;
    return b;
  }
  if (ks + 24 <= Klimit) {
    v8h l0 = *(const v8h*)(W + (long)n * Kphys + ks);
    v8h l1 = *(const v8h*)(W + (long)n * Kphys + ks + 16);
#pragma unroll
    for (int i = 0; i < 8; ++i) { b[i] = l0[i]; b[8 + i] = l1[i]; }
  } else {
#pragma unroll
    for (int i = 0; i < 16; ++i) {
      int kk = ks + (i < 8 ? i : i + 8);
      b[i] = (kk < Klimit) ? W[(long)n * Kphys + kk] : (_Float16)0.f;
    }
  }
  return b;
}

// One dense layer over a 16-row LDS activation tile: dst = act(A @ W^T + bias), f16 out.
// FULLB: N%16==0, Kloop==Kphys==Klimit (no bounds checks on B).
template<bool RELU, bool FULLB>
DEVI void mlp_layer_ws(const _Float16* src, int ldA, int Kloop,
                       const _Float16* W, int Kphys, int Klimit,
                       const float* bias,
                       _Float16* dst, int ldD, int N, int lane) {
  const int m_of = (lane >> 4) << 3;
  const int n    = lane & 15;
  for (int n0 = 0; n0 < N; n0 += 16) {
    v8f acc = wz8();
    for (int k = 0; k < Kloop; k += 32) {
      v16h a = frag_a_lds(src, ldA, k, lane);
      v16h b = FULLB ? frag_b_full(W + (long)n0 * Kphys, Kphys, k, lane)
                     : frag_b_gl(W + (long)n0 * Kphys, Kphys, Klimit, k, lane, N - n0);
      acc = __builtin_amdgcn_wmma_f32_16x16x32_f16(false, a, false, b, (short)0, acc, false, false);
    }
    float bv = bias[n0 + n];
#pragma unroll
    for (int v = 0; v < 8; ++v) {
      float r = acc[v] + bv;
      if (RELU) r = fmaxf(r, 0.f);
      dst[(v + m_of) * ldD + n0 + n] = (_Float16)r;
    }
  }
}

// ---------------------------------------------------------------------------
// Weight convert: f16 copy with BN gain folded (Wh[o][i] = W[o][i]*g[o]/sqrt(1+eps))
// ---------------------------------------------------------------------------
__global__ void rl_convw(const float* __restrict__ W, const float* __restrict__ g,
                         float scale, _Float16* __restrict__ out, int total, int icols) {
  int t = blockIdx.x * 256 + threadIdx.x;
  if (t >= total) return;
  float s = scale;
  if (g) s *= g[t / icols];
  out[t] = (_Float16)(W[t] * s);
}

// ---------------------------------------------------------------------------
// Generic row-batched GEMM: Out = act(A(rows x K) @ W^T + bias), WMMA f16.
// KALIGNED: rows%16==0, Kdim%32==0, Ald%4==0 -> unpredicated b128 A loads.
// NALIGNED: N%16==0 -> unpredicated B loads.
// ---------------------------------------------------------------------------
template<bool RELU, bool KALIGNED, bool NALIGNED>
__global__ void rl_gemm_rows(const float* __restrict__ A, long Abst, int Ald,
                             int rows, int Kdim,
                             const _Float16* __restrict__ W, const float* __restrict__ bias,
                             int N, float* __restrict__ Out, long Obst, int Old) {
  const int wave = threadIdx.x >> 5, lane = threadIdx.x & 31;
  const int tiles_n = (N + 15) >> 4;
  const int tiles_m = (rows + 15) >> 4;
  const int tid = blockIdx.x * (blockDim.x >> 5) + wave;
  if (tid >= tiles_m * tiles_n) return;
  const int mt = tid / tiles_n, nt = tid % tiles_n;
  const float* Ab = A + (long)blockIdx.y * Abst;
  float* Ob = Out + (long)blockIdx.y * Obst;
  const int m = lane & 15, m_of = (lane >> 4) << 3;
  const int row = mt * 16 + m;
  const int Kloop = (Kdim + 31) & ~31;
  v8f acc = wz8();
  for (int k = 0; k < Kloop; k += 32) {
    const int k0 = k + ((lane >> 4) << 3);
    v16h a;
    if (KALIGNED) {
      const float* ap = Ab + (long)row * Ald + k0;
      v4f x0 = *(const v4f*)(ap);
      v4f x1 = *(const v4f*)(ap + 4);
      v4f y0 = *(const v4f*)(ap + 16);
      v4f y1 = *(const v4f*)(ap + 20);
#pragma unroll
      for (int j = 0; j < 4; ++j) {
        a[j]      = (_Float16)x0[j];
        a[4 + j]  = (_Float16)x1[j];
        a[8 + j]  = (_Float16)y0[j];
        a[12 + j] = (_Float16)y1[j];
      }
    } else {
      const bool rok = row < rows;
#pragma unroll
      for (int j = 0; j < 8; ++j) {
        float x0 = (rok && (k0 + j) < Kdim) ? Ab[(long)row * Ald + k0 + j] : 0.f;
        float x1 = (rok && (k0 + 16 + j) < Kdim) ? Ab[(long)row * Ald + k0 + 16 + j] : 0.f;
        a[j] = (_Float16)x0; a[8 + j] = (_Float16)x1;
      }
    }
    v16h b = (NALIGNED && KALIGNED)
                 ? frag_b_full(W + (long)nt * 16 * Kdim, Kdim, k, lane)
                 : frag_b_gl(W + (long)nt * 16 * Kdim, Kdim, Kdim, k, lane, N - nt * 16);
    acc = __builtin_amdgcn_wmma_f32_16x16x32_f16(false, a, false, b, (short)0, acc, false, false);
  }
  const int gn = nt * 16 + (lane & 15);
  float bv = (gn < N) ? bias[gn] : 0.f;
#pragma unroll
  for (int v = 0; v < 8; ++v) {
    int gm = mt * 16 + v + m_of;
    if (gm < rows && gn < N) {
      float r = acc[v] + bv;
      if (RELU) r = fmaxf(r, 0.f);
      Ob[(long)gm * Old + gn] = r;
    }
  }
}

// ---------------------------------------------------------------------------
// Brute-force KNN (K=16) with LDS candidate tiles.
// ---------------------------------------------------------------------------
__global__ void rl_knn(const float* __restrict__ coords, long cbst, int n,
                       int* __restrict__ idxo, long ibst) {
  __shared__ float sx[256], sy[256], sz[256];
  const int tid = threadIdx.x;
  const int q = blockIdx.x * 256 + tid;
  const float* cb = coords + (long)blockIdx.y * cbst;
  const bool valid = q < n;
  float qx = 0.f, qy = 0.f, qz = 0.f;
  if (valid) { qx = cb[(long)q * 3]; qy = cb[(long)q * 3 + 1]; qz = cb[(long)q * 3 + 2]; }
  float bd[16]; int bi[16];
#pragma unroll
  for (int s = 0; s < 16; ++s) { bd[s] = 3.4e38f; bi[s] = 0; }
  for (int t0 = 0; t0 < n; t0 += 256) {
    int j = t0 + tid;
    if (j < n) { sx[tid] = cb[(long)j * 3]; sy[tid] = cb[(long)j * 3 + 1]; sz[tid] = cb[(long)j * 3 + 2]; }
    if (t0 + 256 + tid < n) __builtin_prefetch(cb + (long)(t0 + 256 + tid) * 3, 0, 1);
    __syncthreads();
    const int cnt = min(256, n - t0);
    for (int jj = 0; jj < cnt; ++jj) {
      float dx = qx - sx[jj], dy = qy - sy[jj], dz = qz - sz[jj];
      float d = dx * dx + dy * dy + dz * dz;
      if (d < bd[15]) {
        bd[15] = d; bi[15] = t0 + jj;
#pragma unroll
        for (int s = 15; s > 0; --s)
          if (bd[s] < bd[s - 1]) {
            float td = bd[s]; bd[s] = bd[s - 1]; bd[s - 1] = td;
            int ti = bi[s]; bi[s] = bi[s - 1]; bi[s - 1] = ti;
          }
      }
    }
    __syncthreads();
  }
  if (valid) {
    int* op = idxo + (long)blockIdx.y * ibst + (long)q * 16;
#pragma unroll
    for (int s = 0; s < 16; ++s) op[s] = bi[s];
  }
}

// ---------------------------------------------------------------------------
// Nearest-neighbor interpolation: argmin distance then copy feature row into a
// (possibly column-offset) slice of the output concat buffer.
// ---------------------------------------------------------------------------
__global__ void rl_interp(const float* __restrict__ tc, long tcb, int nt,
                          const float* __restrict__ sc, long scb, int ns,
                          const float* __restrict__ sf, long sfb, int sfld, int C,
                          float* __restrict__ out, long ob, int old_, int coff) {
  __shared__ float sx[256], sy[256], sz[256];
  const int tid = threadIdx.x;
  const int q = blockIdx.x * 256 + tid;
  const int b = blockIdx.y;
  const float* tcbp = tc + (long)b * tcb;
  const float* scbp = sc + (long)b * scb;
  const bool valid = q < nt;
  float qx = 0.f, qy = 0.f, qz = 0.f;
  if (valid) { qx = tcbp[(long)q * 3]; qy = tcbp[(long)q * 3 + 1]; qz = tcbp[(long)q * 3 + 2]; }
  float bd = 3.4e38f; int bidx = 0;
  for (int t0 = 0; t0 < ns; t0 += 256) {
    int j = t0 + tid;
    if (j < ns) { sx[tid] = scbp[(long)j * 3]; sy[tid] = scbp[(long)j * 3 + 1]; sz[tid] = scbp[(long)j * 3 + 2]; }
    __syncthreads();
    const int cnt = min(256, ns - t0);
    for (int jj = 0; jj < cnt; ++jj) {
      float dx = qx - sx[jj], dy = qy - sy[jj], dz = qz - sz[jj];
      float d = dx * dx + dy * dy + dz * dz;
      if (d < bd) { bd = d; bidx = t0 + jj; }
    }
    __syncthreads();
  }
  if (valid) {
    const float* fr = sf + (long)b * sfb + (long)bidx * sfld;
    float* orow = out + (long)b * ob + (long)q * old_ + coff;
    for (int c = 0; c < C; ++c) orow[c] = fr[c];
  }
}

__global__ void rl_addrelu(const float* __restrict__ a, const float* __restrict__ c,
                           float* __restrict__ o, int n) {
  int t = blockIdx.x * 256 + threadIdx.x;
  if (t < n) o[t] = fmaxf(a[t] + c[t], 0.f);
}

// ---------------------------------------------------------------------------
// Fused LFA: one wave per point. M dim of every WMMA = 16 neighbors.
// gather -> LSE(4->C_IN->2C_IN) -> concat -> mlp1(3C_IN->C_OUT/2) ->
// mlp2(->C_OUT) -> attention scores GEMM + softmax(K) + weighted pool.
// ---------------------------------------------------------------------------
template<int C_IN, int C_OUT, int PPB>
__global__ __launch_bounds__(PPB * 32)
void rl_lfa(const float* __restrict__ coords, long cbst,
            const float* __restrict__ feats, long fbst, int fld,
            const int* __restrict__ idx, long ibst,
            const _Float16* __restrict__ Wl0, const float* __restrict__ bl0,
            const _Float16* __restrict__ Wl1, const float* __restrict__ bl1,
            const _Float16* __restrict__ Wm1, const float* __restrict__ bm1,
            const _Float16* __restrict__ Wm2, const float* __restrict__ bm2,
            const _Float16* __restrict__ Wsc,
            float* __restrict__ pooled, long pbst, int n) {
  (void)n;
  extern __shared__ char smemRaw[];
  constexpr int ENC_LD  = 32;
  constexpr int ACAT_LD = 3 * C_IN;
  constexpr int H1_LD   = C_IN;
  constexpr int HALVES  = 16 * ENC_LD + 16 * ACAT_LD + 16 * H1_LD;  // per point
  const int w = threadIdx.x >> 5, lane = threadIdx.x & 31;
  _Float16* base = (_Float16*)smemRaw;
  _Float16* enc  = base + (long)w * HALVES;
  _Float16* acat = enc + 16 * ENC_LD;
  _Float16* h1   = acat + 16 * ACAT_LD;
  int* sidx = (int*)(base + (long)PPB * HALVES) + w * 16;
  const int p = blockIdx.x * PPB + w;
  const int b = blockIdx.y;

  // 1) indices, relative coords, distance -> encIn (16 x 32, zero padded)
  if (lane < 16) {
    int ii = idx[(long)b * ibst + (long)p * 16 + lane];
    sidx[lane] = ii;
    const float* cb = coords + (long)b * cbst;
    float cx = cb[(long)p * 3], cy = cb[(long)p * 3 + 1], cz = cb[(long)p * 3 + 2];
    float rx = cb[(long)ii * 3] - cx, ry = cb[(long)ii * 3 + 1] - cy, rz = cb[(long)ii * 3 + 2] - cz;
    float dd = sqrtf(rx * rx + ry * ry + rz * rz);
    _Float16* er = enc + lane * ENC_LD;
#pragma unroll
    for (int t = 4; t < ENC_LD; ++t) er[t] = (_Float16)0.f;
    er[0] = (_Float16)rx; er[1] = (_Float16)ry; er[2] = (_Float16)rz; er[3] = (_Float16)dd;
  }
  __syncthreads();
  // 2) gather neighbor features into acat[:, 0:C_IN]
  const float* fb = feats + (long)b * fbst;
  for (int e = lane; e < 16 * C_IN; e += 32) {
    int k = e / C_IN, c = e - k * C_IN;
    acat[k * ACAT_LD + c] = (_Float16)fb[(long)sidx[k] * fld + c];
  }
  __syncthreads();
  // 3) LSE layer0: (16x4 pad 32) -> h1 (16 x C_IN)   (short K -> checked B frag)
  mlp_layer_ws<true, false>(enc, ENC_LD, 32, Wl0, 4, 4, bl0, h1, H1_LD, C_IN, lane);
  __syncthreads();
  // 4) LSE layer1: h1 -> acat[:, C_IN : 3C_IN]  (2*C_IN out)
  mlp_layer_ws<true, true>(h1, H1_LD, C_IN, Wl1, C_IN, C_IN, bl1, acat + C_IN, ACAT_LD, 2 * C_IN, lane);
  __syncthreads();
  // 5) mlp1: acat (3C_IN) -> h1 region (C_OUT/2)
  mlp_layer_ws<true, true>(acat, ACAT_LD, 3 * C_IN, Wm1, 3 * C_IN, 3 * C_IN, bm1, h1, C_OUT / 2, C_OUT / 2, lane);
  __syncthreads();
  // 6) mlp2: -> x in acat region (16 x C_OUT, ld C_OUT)
  mlp_layer_ws<true, true>(h1, C_OUT / 2, C_OUT / 2, Wm2, C_OUT / 2, C_OUT / 2, bm2, acat, C_OUT, C_OUT, lane);
  __syncthreads();
  // 7) scores = x @ Ws^T ; softmax over the 16 neighbors ; pooled = sum(score*x)
  const int m_of = (lane >> 4) << 3;
  const int nn = lane & 15;
  for (int n0 = 0; n0 < C_OUT; n0 += 16) {
    v8f acc = wz8();
    for (int k = 0; k < C_OUT; k += 32) {
      v16h a  = frag_a_lds(acat, C_OUT, k, lane);
      v16h bf = frag_b_full(Wsc + (long)n0 * C_OUT, C_OUT, k, lane);
      acc = __builtin_amdgcn_wmma_f32_16x16x32_f16(false, a, false, bf, (short)0, acc, false, false);
    }
    float mx = acc[0];
#pragma unroll
    for (int v = 1; v < 8; ++v) mx = fmaxf(mx, acc[v]);
    mx = fmaxf(mx, __shfl_xor(mx, 16, 32));
    float ex[8]; float sum = 0.f;
#pragma unroll
    for (int v = 0; v < 8; ++v) { ex[v] = __expf(acc[v] - mx); sum += ex[v]; }
    sum += __shfl_xor(sum, 16, 32);
    float isum = 1.f / sum;
    float pn = 0.f;
#pragma unroll
    for (int v = 0; v < 8; ++v)
      pn += ex[v] * isum * (float)acat[(v + m_of) * C_OUT + n0 + nn];
    pn += __shfl_xor(pn, 16, 32);
    if (lane < 16) pooled[(long)b * pbst + (long)p * C_OUT + n0 + nn] = pn;
  }
}

// ---------------------------------------------------------------------------
// Host-side orchestration
// ---------------------------------------------------------------------------
struct LFAW {
  const _Float16 *Wl0, *Wl1, *Wm1, *Wm2, *Wsc, *Wpm;
  const float *bl0, *bl1, *bm1, *bm2, *bpm;
};

static void rl_gemm(const float* A, long Abst, int Ald, int rows, int K,
                    const _Float16* W, const float* bias, int N,
                    float* O, long Obst, bool relu, hipStream_t s) {
  int tm = (rows + 15) / 16, tn = (N + 15) / 16;
  int tiles = tm * tn;
  dim3 grid((tiles + 3) / 4, 2);
  const bool ka = (K % 32 == 0) && (rows % 16 == 0) && (Ald % 4 == 0);
  const bool na = (N % 16 == 0);
  if (ka && na) {
    if (relu) rl_gemm_rows<true,  true,  true><<<grid, 128, 0, s>>>(A, Abst, Ald, rows, K, W, bias, N, O, Obst, N);
    else      rl_gemm_rows<false, true,  true><<<grid, 128, 0, s>>>(A, Abst, Ald, rows, K, W, bias, N, O, Obst, N);
  } else if (ka) {
    if (relu) rl_gemm_rows<true,  true,  false><<<grid, 128, 0, s>>>(A, Abst, Ald, rows, K, W, bias, N, O, Obst, N);
    else      rl_gemm_rows<false, true,  false><<<grid, 128, 0, s>>>(A, Abst, Ald, rows, K, W, bias, N, O, Obst, N);
  } else {
    if (relu) rl_gemm_rows<true,  false, false><<<grid, 128, 0, s>>>(A, Abst, Ald, rows, K, W, bias, N, O, Obst, N);
    else      rl_gemm_rows<false, false, false><<<grid, 128, 0, s>>>(A, Abst, Ald, rows, K, W, bias, N, O, Obst, N);
  }
}

template<int CI, int CO>
static void launch_lfa(const float* coords, long cbst, const float* feats, long fbst, int fld,
                       const int* idx, long ibst, const LFAW& w, float* pooled, long pbst,
                       int n, hipStream_t s) {
  constexpr int PPB = (CI >= 512) ? 2 : 4;
  constexpr int HALVES = 16 * (32 + 4 * CI);
  size_t shmem = (size_t)PPB * HALVES * sizeof(_Float16) + (size_t)PPB * 16 * sizeof(int);
  dim3 grid(n / PPB, 2);
  rl_lfa<CI, CO, PPB><<<grid, dim3(PPB * 32), shmem, s>>>(
      coords, cbst, feats, fbst, fld, idx, ibst,
      w.Wl0, w.bl0, w.Wl1, w.bl1, w.Wm1, w.bm1, w.Wm2, w.bm2, w.Wsc,
      pooled, pbst, n);
}

template<int CI, int CO>
static void run_drb(const float* coords, const float* feats, long fbst, int fld, int n,
                    const LFAW* lw, const _Float16* Wsh, const float* bsh,
                    int* idxbuf, float* bufP, float* bufT1, float* bufT2, float* ident,
                    float* outBuf, hipStream_t s) {
  const long cbst = 8192L * 3;
  rl_knn<<<dim3((n + 255) / 256, 2), 256, 0, s>>>(coords, cbst, n, idxbuf, (long)n * 16);
  rl_gemm(feats, fbst, fld, n, CI, Wsh, bsh, CO, ident, (long)n * CO, false, s);  // shortcut
  launch_lfa<CI, CO>(coords, cbst, feats, fbst, fld, idxbuf, (long)n * 16, lw[0], bufP, (long)n * CO, n, s);
  rl_gemm(bufP, (long)n * CO, CO, n, CO, lw[0].Wpm, lw[0].bpm, CO, bufT1, (long)n * CO, true, s);
  launch_lfa<CO, CO>(coords, cbst, bufT1, (long)n * CO, CO, idxbuf, (long)n * 16, lw[1], bufP, (long)n * CO, n, s);
  rl_gemm(bufP, (long)n * CO, CO, n, CO, lw[1].Wpm, lw[1].bpm, CO, bufT2, (long)n * CO, true, s);
  int tot = 2 * n * CO;
  rl_addrelu<<<(tot + 255) / 256, 256, 0, s>>>(bufT2, ident, outBuf, tot);
}

static void rl_do_interp(const float* tc, long tcb, int nt, const float* sc, long scb, int ns,
                         const float* sf, long sfb, int sfld, int C,
                         float* out, long ob, int old_, int coff, hipStream_t s) {
  dim3 grid((nt + 255) / 256, 2);
  rl_interp<<<grid, 256, 0, s>>>(tc, tcb, nt, sc, scb, ns, sf, sfb, sfld, C, out, ob, old_, coff);
}

extern "C" void kernel_launch(void* const* d_in, const int* in_sizes, int n_in,
                              void* d_out, int out_size, void* d_ws, size_t ws_size,
                              hipStream_t stream) {
  (void)in_sizes; (void)n_in; (void)out_size; (void)ws_size;
  const float* coords   = (const float*)d_in[0];
  const float* features = (const float*)d_in[1];

  char* wsp = (char*)d_ws;
  auto alloc = [&](size_t bytes) -> void* {
    void* r = (void*)wsp;
    wsp += (bytes + 255) & ~(size_t)255;
    return r;
  };

  const float INV_BN = 0.99999500003749971875f;  // 1/sqrt(1 + 1e-5)
  auto cvt = [&](int wi, int gi, int o, int i) -> const _Float16* {
    _Float16* dst = (_Float16*)alloc((size_t)o * i * sizeof(_Float16));
    int total = o * i;
    rl_convw<<<dim3((total + 255) / 256), 256, 0, stream>>>(
        (const float*)d_in[wi], gi >= 0 ? (const float*)d_in[gi] : nullptr,
        gi >= 0 ? INV_BN : 1.0f, dst, total, i);
    return dst;
  };

  // --- parameter leaf indices (jax pytree: dict keys sorted, lists in order) ---
  // [2..13]  dec0..dec3: W,b,g each
  // [14..]   enc0..enc3: lfa1(16) lfa2(16) short(2)
  // then fc_end1 W,b ; fc_end2 W,b ; fc_start W,b
  const int cin[4] = {32, 64, 128, 256};
  const int cou[4] = {64, 128, 256, 512};
  const int dinC[4] = {768, 384, 192, 96};
  const int doutC[4] = {256, 128, 64, 32};

  LFAW LW[4][2];
  const _Float16* WshH[4]; const float* bsh[4];
  int pi = 14;
  for (int e = 0; e < 4; ++e) {
    for (int j = 0; j < 2; ++j) {
      int ci = j ? cou[e] : cin[e];
      int co = cou[e];
      LFAW w;
      w.Wl0 = cvt(pi + 0, pi + 2, ci, 4);          w.bl0 = (const float*)d_in[pi + 1];
      w.Wl1 = cvt(pi + 3, pi + 5, 2 * ci, ci);     w.bl1 = (const float*)d_in[pi + 4];
      w.Wm1 = cvt(pi + 6, pi + 8, co / 2, 3 * ci); w.bm1 = (const float*)d_in[pi + 7];
      w.Wm2 = cvt(pi + 9, pi + 11, co, co / 2);    w.bm2 = (const float*)d_in[pi + 10];
      w.Wsc = cvt(pi + 12, -1, co, co);
      w.Wpm = cvt(pi + 13, pi + 15, co, co);       w.bpm = (const float*)d_in[pi + 14];
      LW[e][j] = w;
      pi += 16;
    }
    WshH[e] = cvt(pi, -1, cou[e], cin[e]);
    bsh[e] = (const float*)d_in[pi + 1];
    pi += 2;
  }
  const _Float16* Wfe1 = cvt(pi, -1, 32, 32); const float* bfe1 = (const float*)d_in[pi + 1]; pi += 2;
  const _Float16* Wfe2 = cvt(pi, -1, 5, 32);  const float* bfe2 = (const float*)d_in[pi + 1]; pi += 2;
  const _Float16* Wfs  = cvt(pi, -1, 32, 6);  const float* bfs  = (const float*)d_in[pi + 1]; pi += 2;
  const _Float16* WdecH[4]; const float* bdec[4];
  for (int i = 0; i < 4; ++i) {
    WdecH[i] = cvt(2 + 3 * i, 2 + 3 * i + 2, doutC[i], dinC[i]);
    bdec[i] = (const float*)d_in[2 + 3 * i + 1];
  }

  // --- activation buffers ---
  float* f0    = (float*)alloc((size_t)2 * 8192 * 32 * 4);
  int*   idxb  = (int*)  alloc((size_t)2 * 8192 * 16 * 4);
  float* ident = (float*)alloc((size_t)2 * 8192 * 64 * 4);
  float* bufP  = (float*)alloc((size_t)2 * 8192 * 64 * 4);
  float* bufT1 = (float*)alloc((size_t)2 * 8192 * 64 * 4);
  float* bufT2 = (float*)alloc((size_t)2 * 8192 * 64 * 4);
  float* e0    = (float*)alloc((size_t)2 * 8192 * 64 * 4);
  float* e1    = (float*)alloc((size_t)2 * 2048 * 128 * 4);
  float* e2    = (float*)alloc((size_t)2 * 512 * 256 * 4);
  float* e3    = (float*)alloc((size_t)2 * 128 * 512 * 4);
  float* cat   = (float*)alloc((size_t)2 * 8192 * 192 * 4);
  float* db0   = (float*)alloc((size_t)2 * 512 * 256 * 4);
  float* db1   = (float*)alloc((size_t)2 * 2048 * 128 * 4);
  float* db2   = (float*)alloc((size_t)2 * 8192 * 64 * 4);
  float* db3   = (float*)alloc((size_t)2 * 8192 * 32 * 4);
  float* db4   = (float*)alloc((size_t)2 * 8192 * 32 * 4);

  const long cbst = 8192L * 3;

  // fc_start: (B,8192,6) -> (B,8192,32)   (K=6 -> unaligned path)
  rl_gemm(features, 8192L * 6, 6, 8192, 6, Wfs, bfs, 32, f0, 8192L * 32, false, stream);

  // encoder (coord levels are prefixes of the input cloud)
  run_drb<32, 64>  (coords, f0, 8192L * 32, 32, 8192, LW[0], WshH[0], bsh[0], idxb, bufP, bufT1, bufT2, ident, e0, stream);
  run_drb<64, 128> (coords, e0, 8192L * 64, 64, 2048, LW[1], WshH[1], bsh[1], idxb, bufP, bufT1, bufT2, ident, e1, stream);
  run_drb<128, 256>(coords, e1, 2048L * 128, 128, 512, LW[2], WshH[2], bsh[2], idxb, bufP, bufT1, bufT2, ident, e2, stream);
  run_drb<256, 512>(coords, e2, 512L * 256, 256, 128, LW[3], WshH[3], bsh[3], idxb, bufP, bufT1, bufT2, ident, e3, stream);

  // decoder: nn-interp up + skip -> concat -> smlp
  // i=0: tgt c2(512), up from (c3,e3:512ch), skip from (c3, e2 prefix:256ch)
  rl_do_interp(coords, cbst, 512, coords, cbst, 128, e3, 128L * 512, 512, 512, cat, 512L * 768, 768, 0, stream);
  rl_do_interp(coords, cbst, 512, coords, cbst, 128, e2, 512L * 256, 256, 256, cat, 512L * 768, 768, 512, stream);
  rl_gemm(cat, 512L * 768, 768, 512, 768, WdecH[0], bdec[0], 256, db0, 512L * 256, true, stream);
  // i=1: tgt c1(2048)
  rl_do_interp(coords, cbst, 2048, coords, cbst, 512, db0, 512L * 256, 256, 256, cat, 2048L * 384, 384, 0, stream);
  rl_do_interp(coords, cbst, 2048, coords, cbst, 512, e1, 2048L * 128, 128, 128, cat, 2048L * 384, 384, 256, stream);
  rl_gemm(cat, 2048L * 384, 384, 2048, 384, WdecH[1], bdec[1], 128, db1, 2048L * 128, true, stream);
  // i=2: tgt c0(8192)
  rl_do_interp(coords, cbst, 8192, coords, cbst, 2048, db1, 2048L * 128, 128, 128, cat, 8192L * 192, 192, 0, stream);
  rl_do_interp(coords, cbst, 8192, coords, cbst, 2048, e0, 8192L * 64, 64, 64, cat, 8192L * 192, 192, 128, stream);
  rl_gemm(cat, 8192L * 192, 192, 8192, 192, WdecH[2], bdec[2], 64, db2, 8192L * 64, true, stream);
  // i=3: tgt c0(8192), sources at full res (up == identity via argmin=self)
  rl_do_interp(coords, cbst, 8192, coords, cbst, 8192, db2, 8192L * 64, 64, 64, cat, 8192L * 96, 96, 0, stream);
  rl_do_interp(coords, cbst, 8192, coords, cbst, 8192, f0, 8192L * 32, 32, 32, cat, 8192L * 96, 96, 64, stream);
  rl_gemm(cat, 8192L * 96, 96, 8192, 96, WdecH[3], bdec[3], 32, db3, 8192L * 32, true, stream);

  // heads
  rl_gemm(db3, 8192L * 32, 32, 8192, 32, Wfe1, bfe1, 32, db4, 8192L * 32, true, stream);
  rl_gemm(db4, 8192L * 32, 32, 8192, 32, Wfe2, bfe2, 5, (float*)d_out, 8192L * 5, false, stream);
}
